// PointTransformer_15857019257413
// MI455X (gfx1250) — matrix operand order repaired and checked by reading, not compiled, not run on hardware
//
#include <hip/hip_runtime.h>

#define B_   32
#define N_   4096
#define S_   512
#define K_   32
#define D_   64
#define EPS_ 1e-5f

typedef _Float16 half_t;
typedef __attribute__((ext_vector_type(16))) _Float16 v16h;
typedef __attribute__((ext_vector_type(8)))  float    v8f;

// ---------------------------------------------------------------------------
// WMMA fragment loaders (wave32, 16x16x32 f16).
// A-matrix 16x32 f16, ISA layout: lane L -> M = L%16, half h = L/16.
//   regs 0..3 hold K = k0 + h*8 + 0..7 ; regs 4..7 hold K = k0 + 16 + h*8 + 0..7
// Source is row-major [M][ld] f16.
__device__ __forceinline__ v16h ldA(const half_t* base, int ld, int m0, int k0, int lane) {
  const half_t* p = base + (size_t)(m0 + (lane & 15)) * ld + k0 + ((lane >> 4) << 3);
  v16h a;
#pragma unroll
  for (int t = 0; t < 8; ++t) a[t] = p[t];
#pragma unroll
  for (int t = 0; t < 8; ++t) a[8 + t] = p[16 + t];
  return a;
}

// B-matrix 32x16 f16, ISA layout: lane L -> N = L%16, half h = L/16.
//   regs 0..3 hold K = k0 + h*8 + 0..7 ; regs 4..7 hold K = k0 + 16 + h*8 + 0..7
// Source stored TRANSPOSED: row-major [N][ld] so the 16 K-values are contiguous.
__device__ __forceinline__ v16h ldBT(const half_t* baseT, int ld, int k0, int n0, int lane) {
  const half_t* p = baseT + (size_t)(n0 + (lane & 15)) * ld + k0 + ((lane >> 4) << 3);
  v16h b;
#pragma unroll
  for (int t = 0; t < 8; ++t) b[t] = p[t];
#pragma unroll
  for (int t = 0; t < 8; ++t) b[8 + t] = p[16 + t];
  return b;
}

__device__ __forceinline__ float gelu_exact(float x) {
  return 0.5f * x * (1.0f + erff(x * 0.70710678118654752f));
}

// wave32 argmax/argmin reductions (tie-break: smaller index wins)
__device__ __forceinline__ void wave_argmax(float& v, int& i) {
#pragma unroll
  for (int m = 16; m > 0; m >>= 1) {
    float ov = __shfl_xor(v, m, 32);
    int   oi = __shfl_xor(i, m, 32);
    if (ov > v || (ov == v && oi < i)) { v = ov; i = oi; }
  }
}
__device__ __forceinline__ void wave_argmin(float& v, int& i) {
#pragma unroll
  for (int m = 16; m > 0; m >>= 1) {
    float ov = __shfl_xor(v, m, 32);
    int   oi = __shfl_xor(i, m, 32);
    if (ov < v || (ov == v && oi < i)) { v = ov; i = oi; }
  }
}

// D = W(64x64) x B(64x32) + bias, optional GELU.
// TR=true: store D transposed f16 [n][64]; TR=false: natural f16 [m][32].
template <bool TR>
__device__ __forceinline__ void gemm64x32(const half_t* W, const float* bias,
                                          const half_t* BT, half_t* dst,
                                          int lane, bool dogelu) {
#pragma unroll
  for (int mi = 0; mi < 4; ++mi)
#pragma unroll
    for (int nj = 0; nj < 2; ++nj) {
      v8f c = {};
#pragma unroll
      for (int kk = 0; kk < 2; ++kk) {
        v16h a  = ldA(W, 64, mi * 16, kk * 32, lane);
        v16h bb = ldBT(BT, 64, kk * 32, nj * 16, lane);
        c = __builtin_amdgcn_wmma_f32_16x16x32_f16(false, a, false, bb, (short)0, c,
                                                   false, false);
      }
      int hi = lane >> 4, nn = (nj << 4) + (lane & 15);
#pragma unroll
      for (int r = 0; r < 8; ++r) {
        int mm = (mi << 4) + (hi << 3) + r;  // C layout: lanes0-15 M=r, lanes16-31 M=8+r
        float val = c[r] + bias[mm];
        if (dogelu) val = gelu_exact(val);
        if (TR) dst[(size_t)nn * 64 + mm] = (half_t)val;
        else    dst[(size_t)mm * 32 + nn] = (half_t)val;
      }
    }
}

// ---------------------------------------------------------------------------
// Kernel 0: weight fp32->fp16 conversion + fused BN scale/shift.
__global__ __launch_bounds__(256) void k0_prep(
    const float* wq, const float* wk, const float* wv,
    const float* wg1, const float* wg2,
    const float* b_alpha, const float* gamma, const float* beta,
    const float* mean, const float* var,
    half_t* W16, float* bn_s, float* bn_t) {
  int i = blockIdx.x * 256 + threadIdx.x;
  if (i < 4096) {
    W16[i]             = (half_t)wq[i];
    W16[4096 + i]      = (half_t)wk[i];
    W16[2 * 4096 + i]  = (half_t)wv[i];
    W16[3 * 4096 + i]  = (half_t)wg1[i];
    W16[4 * 4096 + i]  = (half_t)wg2[i];
  }
  if (i < 64) {
    float sc = gamma[i] * rsqrtf(var[i] + EPS_);
    bn_s[i] = sc;
    bn_t[i] = (b_alpha[i] - mean[i]) * sc + beta[i];
  }
}

// ---------------------------------------------------------------------------
// Kernel 1: farthest point sampling — one block per batch, sequential S steps.
// Points cached in LDS (48 KB); per-step argmax = wave shuffle + 8-way combine.
__global__ __launch_bounds__(256) void k1_fps(const float* x, int* fidx, float* cent) {
  __shared__ float xs[N_ * 3];   // 48 KB point cache
  __shared__ float dists[N_];    // 16 KB
  __shared__ float rv[8];
  __shared__ int   ri[8];
  __shared__ int   s_last;
  const int b = blockIdx.x, tid = threadIdx.x;
  const int wv = tid >> 5, ln = tid & 31;
  for (int t = tid; t < N_ * 3; t += 256) xs[t] = x[(size_t)b * N_ * 3 + t];
  for (int t = tid; t < N_; t += 256) dists[t] = 1e10f;
  if (tid == 0) s_last = 0;
  __syncthreads();
  for (int s = 0; s < S_; ++s) {
    const int last = s_last;
    float cx = xs[last * 3], cy = xs[last * 3 + 1], cz = xs[last * 3 + 2];
    if (tid == 0) {
      fidx[b * S_ + s] = last;
      float* cc = cent + ((size_t)b * S_ + s) * 3;
      cc[0] = cx; cc[1] = cy; cc[2] = cz;
    }
    float best = -1.0f; int bi = 0x7fffffff;
    for (int t = tid; t < N_; t += 256) {
      float dx = xs[t * 3] - cx, dy = xs[t * 3 + 1] - cy, dz = xs[t * 3 + 2] - cz;
      float d  = dx * dx + dy * dy + dz * dz;
      float dm = fminf(dists[t], d);
      dists[t] = dm;
      if (dm > best || (dm == best && t < bi)) { best = dm; bi = t; }
    }
    wave_argmax(best, bi);
    if (ln == 0) { rv[wv] = best; ri[wv] = bi; }
    __syncthreads();
    if (tid == 0) {
      float bv = rv[0]; int bx = ri[0];
#pragma unroll
      for (int w = 1; w < 8; ++w)
        if (rv[w] > bv || (rv[w] == bv && ri[w] < bx)) { bv = rv[w]; bx = ri[w]; }
      s_last = bx;
    }
    __syncthreads();
  }
}

// ---------------------------------------------------------------------------
// Kernel 2: KNN (32 smallest d2) — one block per centroid, iterative argmin
// with wave shuffle reductions.
__global__ __launch_bounds__(256) void k2_knn(const float* x, const float* cent, int* knn) {
  __shared__ float dist[N_];
  __shared__ float rv[8];
  __shared__ int   ri[8];
  const int grp = blockIdx.x;
  const int b = grp / S_, tid = threadIdx.x;
  const int wv = tid >> 5, ln = tid & 31;
  const float* c = cent + (size_t)grp * 3;
  float cx = c[0], cy = c[1], cz = c[2];
  float c2 = cx * cx + cy * cy + cz * cz;
  for (int t = tid; t < N_; t += 256) {
    const float* p = x + ((size_t)b * N_ + t) * 3;
    float p2 = p[0] * p[0] + p[1] * p[1] + p[2] * p[2];
    dist[t] = c2 + p2 - 2.0f * (cx * p[0] + cy * p[1] + cz * p[2]);
  }
  __syncthreads();
  for (int kk = 0; kk < K_; ++kk) {
    float best = 3.3e38f; int bi = 0x7fffffff;
    for (int t = tid; t < N_; t += 256) {
      float d = dist[t];
      if (d < best || (d == best && t < bi)) { best = d; bi = t; }
    }
    wave_argmin(best, bi);
    if (ln == 0) { rv[wv] = best; ri[wv] = bi; }
    __syncthreads();
    if (tid == 0) {
      float bv = rv[0]; int bx = ri[0];
#pragma unroll
      for (int w = 1; w < 8; ++w)
        if (rv[w] < bv || (rv[w] == bv && ri[w] < bx)) { bv = rv[w]; bx = ri[w]; }
      knn[(size_t)grp * K_ + kk] = bx;
      dist[bx] = 3.4e38f;
    }
    __syncthreads();
  }
}

// ---------------------------------------------------------------------------
// Kernel 3: per-group transformer — one wave per group, 4 waves per block.
// LDS per wave (32 KB): F0 f32[64][32] (h, later out), F1 f32[64][32] (logits, res),
// HT f16[32][64] (h^T, later attn[32][32]), QT f16[32][64] (q^T, later t^T),
// KT f16[32][64] (k^T, later u^T), VV f16[64][32] (v).
__global__ __launch_bounds__(128) void k3_group(
    const float* x, const float* cent, const int* knn,
    const float* w_alpha, const float* bn_s, const float* bn_t,
    const half_t* W16,
    const float* bq, const float* bk, const float* bv,
    const float* bg1, const float* bg2,
    float* out) {
  __shared__ __align__(16) unsigned char smem[4][32768];
  const int wave = threadIdx.x >> 5;
  const int lane = threadIdx.x & 31;
  const int grp  = blockIdx.x * 4 + wave;
  const int b    = grp / S_;

  float*  F0 = (float*)(smem[wave]);
  float*  F1 = (float*)(smem[wave] + 8192);
  half_t* HT = (half_t*)(smem[wave] + 16384);
  half_t* QT = (half_t*)(smem[wave] + 20480);
  half_t* KT = (half_t*)(smem[wave] + 24576);
  half_t* VV = (half_t*)(smem[wave] + 28672);

  const half_t* Wq  = W16;
  const half_t* Wk  = W16 + 4096;
  const half_t* Wv  = W16 + 2 * 4096;
  const half_t* Wg1 = W16 + 3 * 4096;
  const half_t* Wg2 = W16 + 4 * 4096;

  // ---- gather: lane = group point index k ----
  int idx = knn[(size_t)grp * K_ + lane];
  float cx = cent[(size_t)grp * 3 + 0];
  float cy = cent[(size_t)grp * 3 + 1];
  float cz = cent[(size_t)grp * 3 + 2];
  const float* xp = x + ((size_t)b * N_ + idx) * 3;
  float p0 = xp[0] - cx, p1 = xp[1] - cy, p2 = xp[2] - cz;

  // ---- h = GELU(BN(Walpha * g)) : f32 in F0[d][k], f16 transposed in HT[k][d] ----
  for (int d = 0; d < D_; ++d) {
    float hv = w_alpha[d * 3 + 0] * p0 + w_alpha[d * 3 + 1] * p1 + w_alpha[d * 3 + 2] * p2;
    hv = hv * bn_s[d] + bn_t[d];
    hv = gelu_exact(hv);
    F0[d * 32 + lane] = hv;
    HT[lane * 64 + d] = (half_t)hv;
  }

  // ---- q, k, v = W * h + b (f16 WMMA, f32 accumulate) ----
  gemm64x32<true >(Wq, bq, HT, QT, lane, false);  // q^T [i][d]
  gemm64x32<true >(Wk, bk, HT, KT, lane, false);  // k^T [j][d]
  gemm64x32<false>(Wv, bv, HT, VV, lane, false);  // v   [d][j]

  // ---- logits = (q^T k) * D^-1/2 -> F1 as f32 [i][j] ----
#pragma unroll
  for (int mi = 0; mi < 2; ++mi)
#pragma unroll
    for (int nj = 0; nj < 2; ++nj) {
      v8f c = {};
#pragma unroll
      for (int kk = 0; kk < 2; ++kk) {
        v16h a  = ldA(QT, 64, mi * 16, kk * 32, lane);
        v16h bb = ldBT(KT, 64, kk * 32, nj * 16, lane);
        c = __builtin_amdgcn_wmma_f32_16x16x32_f16(false, a, false, bb, (short)0, c,
                                                   false, false);
      }
      int hi = lane >> 4, nn = (nj << 4) + (lane & 15);
#pragma unroll
      for (int r = 0; r < 8; ++r)
        F1[((mi << 4) + (hi << 3) + r) * 32 + nn] = c[r] * 0.125f;
    }

  // ---- softmax over j, per row i = lane ; attn f16 [i][j] into HT (reuse) ----
  half_t* AT = HT;
  {
    float row[32];
    float m = -3.4e38f;
#pragma unroll
    for (int j = 0; j < 32; ++j) { row[j] = F1[lane * 32 + j]; m = fmaxf(m, row[j]); }
    float sum = 0.0f;
#pragma unroll
    for (int j = 0; j < 32; ++j) { row[j] = expf(row[j] - m); sum += row[j]; }
    float inv = 1.0f / sum;
#pragma unroll
    for (int j = 0; j < 32; ++j) AT[lane * 32 + j] = (half_t)(row[j] * inv);
  }

  // ---- res[d][i] = sum_j v[d][j] attn[i][j] ; also t = res + h -> QT (t^T) ----
#pragma unroll
  for (int mi = 0; mi < 4; ++mi)
#pragma unroll
    for (int nj = 0; nj < 2; ++nj) {
      v16h a  = ldA(VV, 32, mi * 16, 0, lane);       // A = v, K=32
      v16h bb = ldBT(AT, 32, 0, nj * 16, lane);      // B = attn^T via [i][j] storage
      v8f c = {};
      c = __builtin_amdgcn_wmma_f32_16x16x32_f16(false, a, false, bb, (short)0, c,
                                                 false, false);
      int hi = lane >> 4, nn = (nj << 4) + (lane & 15);
#pragma unroll
      for (int r = 0; r < 8; ++r) {
        int mm = (mi << 4) + (hi << 3) + r;
        float rvv = c[r];
        F1[mm * 32 + nn] = rvv;                      // res (f32)
        float tv = rvv + F0[mm * 32 + nn];           // t = res + h
        QT[nn * 64 + mm] = (half_t)tv;               // t^T
      }
    }

  // ---- u = GELU(Wg1 * t + b1) -> KT (u^T) ----
  gemm64x32<true>(Wg1, bg1, QT, KT, lane, true);

  // ---- out = Wg2 * u + b2 + res -> F0 f32 [d][k] ----
#pragma unroll
  for (int mi = 0; mi < 4; ++mi)
#pragma unroll
    for (int nj = 0; nj < 2; ++nj) {
      v8f c = {};
#pragma unroll
      for (int kk = 0; kk < 2; ++kk) {
        v16h a  = ldA(Wg2, 64, mi * 16, kk * 32, lane);
        v16h bb = ldBT(KT, 64, kk * 32, nj * 16, lane);
        c = __builtin_amdgcn_wmma_f32_16x16x32_f16(false, a, false, bb, (short)0, c,
                                                   false, false);
      }
      int hi = lane >> 4, nn = (nj << 4) + (lane & 15);
#pragma unroll
      for (int r = 0; r < 8; ++r) {
        int mm = (mi << 4) + (hi << 3) + r;
        F0[mm * 32 + nn] = c[r] + bg2[mm] + F1[mm * 32 + nn];
      }
    }

  // ---- max over k, write [centroid(3) | patches(64)] ----
  const size_t base = (size_t)grp * 67;
  if (lane == 0) { out[base] = cx; out[base + 1] = cy; out[base + 2] = cz; }
#pragma unroll
  for (int t = 0; t < 2; ++t) {
    int d = lane + t * 32;
    float m = F0[d * 32];
#pragma unroll
    for (int j = 1; j < 32; ++j) m = fmaxf(m, F0[d * 32 + j]);
    out[base + 3 + d] = m;
  }
}

// ---------------------------------------------------------------------------
extern "C" void kernel_launch(void* const* d_in, const int* in_sizes, int n_in,
                              void* d_out, int out_size, void* d_ws, size_t ws_size,
                              hipStream_t stream) {
  (void)in_sizes; (void)n_in; (void)out_size; (void)ws_size;
  const float* x        = (const float*)d_in[0];
  const float* w_alpha  = (const float*)d_in[1];
  const float* b_alpha  = (const float*)d_in[2];
  const float* bn_gamma = (const float*)d_in[3];
  const float* bn_beta  = (const float*)d_in[4];
  const float* bn_mean  = (const float*)d_in[5];
  const float* bn_var   = (const float*)d_in[6];
  const float* w_q  = (const float*)d_in[7];
  const float* b_q  = (const float*)d_in[8];
  const float* w_k  = (const float*)d_in[9];
  const float* b_k  = (const float*)d_in[10];
  const float* w_v  = (const float*)d_in[11];
  const float* b_v  = (const float*)d_in[12];
  const float* w_g1 = (const float*)d_in[13];
  const float* b_g1 = (const float*)d_in[14];
  const float* w_g2 = (const float*)d_in[15];
  const float* b_g2 = (const float*)d_in[16];
  float* out = (float*)d_out;

  unsigned char* ws = (unsigned char*)d_ws;
  int*    fidx = (int*)ws;                               //  32*512 ints      (64 KB)
  float*  cent = (float*)(ws + 65536);                   //  32*512*3 floats  (192 KB)
  int*    knn  = (int*)(ws + 262144);                    //  16384*32 ints    (2 MB)
  half_t* W16  = (half_t*)(ws + 2359296);                //  5*4096 halves    (40 KB)
  float*  bn_s = (float*)(ws + 2400384);                 //  64 floats
  float*  bn_t = (float*)(ws + 2400640);                 //  64 floats

  k0_prep<<<16, 256, 0, stream>>>(w_q, w_k, w_v, w_g1, w_g2,
                                  b_alpha, bn_gamma, bn_beta, bn_mean, bn_var,
                                  W16, bn_s, bn_t);
  k1_fps<<<B_, 256, 0, stream>>>(x, fidx, cent);
  k2_knn<<<B_ * S_, 256, 0, stream>>>(x, cent, knn);
  k3_group<<<(B_ * S_) / 4, 128, 0, stream>>>(x, cent, knn,
                                              w_alpha, bn_s, bn_t, W16,
                                              b_q, b_k, b_v, b_g1, b_g2, out);
}
